// GNN_23287312679085
// MI455X (gfx1250) — matrix operand order repaired and checked by reading, not compiled, not run on hardware
//
#include <hip/hip_runtime.h>

typedef __attribute__((ext_vector_type(16))) _Float16 v16h;
typedef __attribute__((ext_vector_type(8)))  _Float16 v8h;
typedef __attribute__((ext_vector_type(2)))  _Float16 v2h;
typedef __attribute__((ext_vector_type(8)))  float    v8f;

#define ASTR 132    // sA row stride (halves): 66 words -> conflict-free row-parallel writes
#define HSTR 128    // sH row stride (halves): weights, contiguous b128 reads
#define TSTR 136    // sHT row stride (halves): 68 words -> bank = 4n, conflict-free b128 stores

// ---- A fragment: 16x32 f16, rows m0..m0+15, K window [k0,k0+32) ----
// Layout: lanes 0-15 / 16-31 both hold M = lane&15;
// VGPR v: K = 2v (+16 for hi half-lanes), matching load_fragB.
__device__ __forceinline__ v16h load_fragA(const _Float16* A, int stride,
                                           int m0, int k0, int lane) {
  const int m  = m0 + (lane & 15);
  const int hi = (lane >> 4) << 4;          // 0 or 16
  const _Float16* row = A + m * stride;
  v16h a;
#pragma unroll
  for (int v = 0; v < 8; ++v) {
    int k = k0 + 2 * v + hi;
    a[2 * v]     = row[k];
    a[2 * v + 1] = row[k + 1];
  }
  return a;
}

// ---- B fragment: 32x16 f16, cols n0..n0+15, K window [k0,k0+32) ----
// B stored row-major as [n][k] (B(k,n) = B[n*stride + k]); 32 contiguous
// bytes per lane -> two ds_load_b128.
__device__ __forceinline__ v16h load_fragB(const _Float16* B, int stride,
                                           int n0, int k0, int lane) {
  const int n  = n0 + (lane & 15);
  const int kb = k0 + ((lane >> 4) << 4);
  const _Float16* col = B + n * stride + kb;
  v16h b;
#pragma unroll
  for (int v = 0; v < 8; ++v) {
    b[2 * v]     = col[2 * v];
    b[2 * v + 1] = col[2 * v + 1];
  }
  return b;
}

// 16x128 output strip (rows m0..m0+15, all 128 cols), K=128, f32 accumulate.
__device__ __forceinline__ void gemm_strip(const _Float16* A, int strideA,
                                           const _Float16* B, int strideB,
                                           int m0, int lane, v8f acc[8]) {
#pragma unroll
  for (int kt = 0; kt < 4; ++kt) {
    v16h a = load_fragA(A, strideA, m0, kt * 32, lane);
#pragma unroll
    for (int nt = 0; nt < 8; ++nt) {
      v16h b = load_fragB(B, strideB, nt * 16, kt * 32, lane);
      acc[nt] = __builtin_amdgcn_wmma_f32_16x16x32_f16(
          false, a, false, b, (short)0, acc[nt], false, false);
    }
  }
}

// Transposed strip store: C/D layout gives each lane 8 consecutive rows
// (M = mo..mo+7) of column n -> one 16B ds_store_b128 per tile per lane.
__device__ __forceinline__ void store_strip_T(_Float16* DT, int tstride,
                                              int m0, int lane, const v8f acc[8]) {
  const int n_lo = lane & 15;
  const int mo   = (lane >> 4) << 3;
#pragma unroll
  for (int nt = 0; nt < 8; ++nt) {
    const int n = nt * 16 + n_lo;
    v8h p;
#pragma unroll
    for (int v = 0; v < 8; ++v) p[v] = (_Float16)acc[nt][v];
    *(v8h*)(DT + n * tstride + m0 + mo) = p;
  }
}

// Row-major strip store (+bias): used for g0, which must be an A operand.
__device__ __forceinline__ void store_strip_R(_Float16* D, int dstride,
                                              int m0, int lane, const v8f acc[8],
                                              const float* bias) {
  const int n_lo = lane & 15;
  const int mo   = (lane >> 4) << 3;
#pragma unroll
  for (int nt = 0; nt < 8; ++nt) {
    const int n = nt * 16 + n_lo;
    const float bv = bias[n];
#pragma unroll
    for (int v = 0; v < 8; ++v)
      D[(m0 + mo + v) * dstride + n] = (_Float16)(acc[nt][v] + bv);
  }
}

// e_src/e_dst from the TRANSPOSED features: per iteration all 256 lanes read
// contiguous halves of one sHT row -> conflict-free, vec[d] is a broadcast.
__device__ __forceinline__ void compute_e(const _Float16* HT, const float* vsrc,
                                          const float* vdst, float* es, float* ed,
                                          int tid) {
  const int c = tid & 127;
  const float* vec = (tid < 128) ? vsrc : vdst;
  float s = 0.0f;
#pragma unroll 4
  for (int d = 0; d < 128; ++d) s += (float)HT[d * TSTR + c] * vec[d];
  ((tid < 128) ? es : ed)[c] = s;
}

__device__ __forceinline__ float lrelu02(float v) { return v > 0.0f ? v : 0.2f * v; }

// Row-wise softmax of leaky_relu(ed[i]+es[j], 0.2) into f16 attn matrix
// (threads 0-127). Packed 32-bit writes; ASTR=132 -> bank = 2*lane, no conflicts.
__device__ __forceinline__ void compute_attn(_Float16* Amat, const float* es,
                                             const float* ed, int i) {
  const float edi = ed[i];
  float mx = -1e30f;
  for (int j = 0; j < 128; ++j) mx = fmaxf(mx, lrelu02(edi + es[j]));
  float ssum = 0.0f;
  for (int j = 0; j < 128; ++j) ssum += __expf(lrelu02(edi + es[j]) - mx);
  const float inv = 1.0f / ssum;
  _Float16* arow = Amat + i * ASTR;
  for (int jj = 0; jj < 64; ++jj) {
    const int j0 = 2 * jj;
    v2h p;
    p[0] = (_Float16)(__expf(lrelu02(edi + es[j0])     - mx) * inv);
    p[1] = (_Float16)(__expf(lrelu02(edi + es[j0 + 1]) - mx) * inv);
    *(v2h*)(arow + j0) = p;
  }
}

__global__ __launch_bounds__(256)
void gat_block_kernel(const float* __restrict__ xs, const float* __restrict__ pos_enc,
                      const float* __restrict__ w0, const float* __restrict__ asrc0,
                      const float* __restrict__ adst0, const float* __restrict__ bias0,
                      const float* __restrict__ w1, const float* __restrict__ asrc1,
                      const float* __restrict__ adst1, const float* __restrict__ bias1,
                      const float* __restrict__ fw, float* __restrict__ out) {
  __shared__ _Float16 sA [128 * ASTR];   // X -> attn0 -> g0 -> attn1  (A operands)
  __shared__ _Float16 sH [128 * HSTR];   // W0 -> W1                   (weights, [d][k])
  __shared__ _Float16 sHT[128 * TSTR];   // h0^T -> h1^T               (features, [d][c])
  __shared__ float sEs[128], sEd[128], sSum[128];
  __shared__ float sVs0[128], sVd0[128], sB0[128], sVs1[128], sVd1[128], sB1[128];

  const int tid  = threadIdx.x;
  const int lane = tid & 31;
  const int wave = tid >> 5;
  const int m0   = wave * 16;
  const int blk  = blockIdx.x;           // b*128 + r
  const int b    = blk >> 7;

  const float* w0b = w0 + (size_t)b * 128 * 128;
  const float* w1b = w1 + (size_t)b * 128 * 128;

  // Prefetch W1 (needed mid-kernel): 256 threads x 256B covers the 64KB matrix.
  __builtin_prefetch((const void*)(w1b + (size_t)tid * 64), 0, 1);

  // ---- stage X (f16) into sA, W0 (f16, [d][k]) into sH, all attn vectors ----
  {
    const float* xs_row = xs + (size_t)blk * 128;
    const float* pe     = pos_enc + (size_t)b * 128 * 127;
    for (int idx = tid; idx < 128 * 64; idx += 256) {
      const int c = idx >> 6, kp = (idx & 63) * 2;
      float x0 = (kp == 0) ? xs_row[c] : pe[c * 127 + kp - 1];
      float x1 = pe[c * 127 + kp];
      v2h px; px[0] = (_Float16)x0; px[1] = (_Float16)x1;
      *(v2h*)(sA + c * ASTR + kp) = px;
      v2h pw; pw[0] = (_Float16)w0b[c * 128 + kp]; pw[1] = (_Float16)w0b[c * 128 + kp + 1];
      *(v2h*)(sH + c * HSTR + kp) = pw;
    }
    if (tid < 128) {
      sVs0[tid] = asrc0[b * 128 + tid];
      sVd0[tid] = adst0[b * 128 + tid];
      sB0[tid]  = bias0[b * 128 + tid];
      sVs1[tid] = asrc1[b * 128 + tid];
      sVd1[tid] = adst1[b * 128 + tid];
      sB1[tid]  = bias1[b * 128 + tid];
    }
  }
  __syncthreads();

  v8f acc[8];

  // ---- GEMM0: h0 = X @ W0^T -> sHT (transposed only; no conflicting readers) ----
#pragma unroll
  for (int i = 0; i < 8; ++i) acc[i] = (v8f)0.0f;
  gemm_strip(sA, ASTR, sH, HSTR, m0, lane, acc);
  store_strip_T(sHT, TSTR, m0, lane, acc);
  __syncthreads();                                      // h0^T visible; all past GEMM0

  // ---- attention 0 ----
  compute_e(sHT, sVs0, sVd0, sEs, sEd, tid);
  __syncthreads();
  if (tid < 128) compute_attn(sA, sEs, sEd, tid);       // sA <- attn0
  __syncthreads();

  // ---- GEMM1: g0 = attn0 @ h0 + bias0 (B = h0^T); then stage W1 over W0 ----
#pragma unroll
  for (int i = 0; i < 8; ++i) acc[i] = (v8f)0.0f;
  gemm_strip(sA, ASTR, sHT, TSTR, m0, lane, acc);
  store_strip_R(sA, ASTR, m0, lane, acc, sB0);          // own A-rows: race-free
  for (int idx = tid; idx < 128 * 64; idx += 256) {     // sH <- W1 (no wave reads sH now)
    const int d = idx >> 6, kp = (idx & 63) * 2;
    v2h pw; pw[0] = (_Float16)w1b[d * 128 + kp]; pw[1] = (_Float16)w1b[d * 128 + kp + 1];
    *(v2h*)(sH + d * HSTR + kp) = pw;
  }
  __syncthreads();                                      // g0 + W1 visible; all past GEMM1

  // ---- GEMM2: h1 = g0 @ W1^T -> sHT ----
#pragma unroll
  for (int i = 0; i < 8; ++i) acc[i] = (v8f)0.0f;
  gemm_strip(sA, ASTR, sH, HSTR, m0, lane, acc);
  store_strip_T(sHT, TSTR, m0, lane, acc);              // h0^T dead (all past GEMM1)
  __syncthreads();

  // ---- attention 1 ----
  compute_e(sHT, sVs1, sVd1, sEs, sEd, tid);
  __syncthreads();
  if (tid < 128) { sSum[tid] = 0.0f; compute_attn(sA, sEs, sEd, tid); }
  __syncthreads();

  // ---- GEMM3: agg1 = attn1 @ h1 (B = h1^T); fold column sums directly ----
#pragma unroll
  for (int i = 0; i < 8; ++i) acc[i] = (v8f)0.0f;
  gemm_strip(sA, ASTR, sHT, TSTR, m0, lane, acc);
  {
    const int n_lo = lane & 15;
#pragma unroll
    for (int nt = 0; nt < 8; ++nt) {
      float s = 0.0f;
#pragma unroll
      for (int v = 0; v < 8; ++v) s += acc[nt][v];
      atomicAdd(&sSum[nt * 16 + n_lo], s);              // ds_add_f32
    }
  }
  __syncthreads();

  // ---- final: out[b,r,o] = fw[b,o,:] . (colsum + 128*bias1) ----
  if (tid < 2) {
    const float* fwb = fw + (size_t)b * 2 * 128 + (size_t)tid * 128;
    float o = 0.0f;
    for (int d = 0; d < 128; ++d) o += fwb[d] * (sSum[d] + 128.0f * sB1[d]);
    out[(size_t)blk * 2 + tid] = o;
  }
}

extern "C" void kernel_launch(void* const* d_in, const int* in_sizes, int n_in,
                              void* d_out, int out_size, void* d_ws, size_t ws_size,
                              hipStream_t stream) {
  (void)in_sizes; (void)n_in; (void)d_ws; (void)ws_size; (void)out_size;
  const float* xs  = (const float*)d_in[0];
  const float* pe  = (const float*)d_in[1];
  const float* w0  = (const float*)d_in[2];
  const float* as0 = (const float*)d_in[3];
  const float* ad0 = (const float*)d_in[4];
  const float* b0  = (const float*)d_in[5];
  const float* w1  = (const float*)d_in[6];
  const float* as1 = (const float*)d_in[7];
  const float* ad1 = (const float*)d_in[8];
  const float* b1  = (const float*)d_in[9];
  const float* fwp = (const float*)d_in[10];
  gat_block_kernel<<<dim3(16 * 128), dim3(256), 0, stream>>>(
      xs, pe, w0, as0, ad0, b0, w1, as1, ad1, b1, fwp, (float*)d_out);
}
